// CausalCrisisLoss_79250736546292
// MI455X (gfx1250) — matrix operand order repaired and checked by reading, not compiled, not run on hardware
//
#include <hip/hip_runtime.h>
#include <hip/hip_bf16.h>
#include <math.h>

// CDNA5 (gfx1250) wave32 WMMA types
typedef __attribute__((ext_vector_type(16))) __bf16 v16bf;
typedef __attribute__((ext_vector_type(8)))  float  v8f;

#define SMOOTH 0.1f
#define NGROUPS 16   // 4 labels x 4 domains

// ---------- helpers ----------
__device__ inline unsigned short f2bf(float f) {
    // round-to-nearest-even float -> bf16 (bit trick)
    unsigned u = __float_as_uint(f);
    unsigned r = u + 0x7FFFu + ((u >> 16) & 1u);
    return (unsigned short)(r >> 16);
}
__device__ inline float bf2f(unsigned short h) {
    return __uint_as_float(((unsigned)h) << 16);
}
__device__ inline void block_reduce_add(float v, float* slot, float* sm) {
    int t = threadIdx.x;
    sm[t] = v; __syncthreads();
    for (int s = blockDim.x >> 1; s > 0; s >>= 1) {
        if (t < s) sm[t] += sm[t + s];
        __syncthreads();
    }
    if (t == 0) atomicAdd(slot, sm[0]);
    __syncthreads();
}

// ---------- zero init ----------
__global__ void zero_kernel(float* p, int n) {
    int i = blockIdx.x * blockDim.x + threadIdx.x;
    if (i < n) p[i] = 0.0f;
}

// ---------- per-row classification losses + group ids ----------
__device__ inline float focal_row(const float* lg, int C, int tgt) {
    float mx = -1e30f;
    for (int c = 0; c < C; ++c) mx = fmaxf(mx, lg[c]);
    float se = 0.0f;
    for (int c = 0; c < C; ++c) se += __expf(lg[c] - mx);
    float lse = __logf(se) + mx;
    float ce = 0.0f;
    float sm_off = SMOOTH / (float)C;
    for (int c = 0; c < C; ++c) {
        float lp = lg[c] - lse;
        float w  = (c == tgt ? (1.0f - SMOOTH) : 0.0f) + sm_off;
        ce -= w * lp;
    }
    float pt = __expf(lg[tgt] - lse);
    float om = 1.0f - pt;
    return ce * om * om;              // FOCAL_GAMMA = 2
}
__device__ inline float ce_row(const float* lg, int C, int tgt) {
    float mx = -1e30f;
    for (int c = 0; c < C; ++c) mx = fmaxf(mx, lg[c]);
    float se = 0.0f;
    for (int c = 0; c < C; ++c) se += __expf(lg[c] - mx);
    return (__logf(se) + mx) - lg[tgt];
}

__global__ void row_losses_kernel(const float* lg1, const int* lb1,
                                  const float* lg2, const int* lb2,
                                  const float* lg3, const int* lb3,
                                  const float* vsv, const float* vst,
                                  const int* dom, const unsigned char* mask,
                                  float* acc, float* ncount, int* gid, int B) {
    __shared__ float sm[256];
    int i = blockIdx.x * blockDim.x + threadIdx.x;
    float mf = 0.f, f1 = 0.f, f2 = 0.f, f3 = 0.f, c1 = 0.f, c2 = 0.f;
    if (i < B) {
        mf = mask[i] ? 1.0f : 0.0f;
        f1 = focal_row(lg1 + (size_t)i * 4, 4, lb1[i]) * mf;
        f2 = focal_row(lg2 + (size_t)i * 6, 6, lb2[i]) * mf;
        f3 = focal_row(lg3 + (size_t)i * 3, 3, lb3[i]) * mf;
        c1 = ce_row(vsv + (size_t)i * 4, 4, dom[i]) * mf;
        c2 = ce_row(vst + (size_t)i * 4, 4, dom[i]) * mf;
        int lab = lb1[i];
        int g = (mf > 0.f && lab >= 0 && lab < 4) ? (lab * 4 + dom[i]) : -1;
        gid[i] = g;
        if (g >= 0) atomicAdd(&ncount[g], 1.0f);
    }
    block_reduce_add(mf, &acc[0], sm);
    block_reduce_add(f1, &acc[1], sm);
    block_reduce_add(f2, &acc[2], sm);
    block_reduce_add(f3, &acc[3], sm);
    block_reduce_add(c1, &acc[4], sm);
    block_reduce_add(c2, &acc[5], sm);
}

// ---------- reconstruction MSE partial sums ----------
__global__ void recon_kernel(const float* ir, const float* io,
                             const float* tr_, const float* to,
                             float* acc, int N) {
    __shared__ float sm[256];
    float s1 = 0.f, s2 = 0.f;
    for (int i = blockIdx.x * blockDim.x + threadIdx.x; i < N;
         i += gridDim.x * blockDim.x) {
        float d1 = ir[i] - io[i];  s1 += d1 * d1;
        float d2 = tr_[i] - to[i]; s2 += d2 * d2;
    }
    block_reduce_add(s1, &acc[6], sm);
    block_reduce_add(s2, &acc[7], sm);
}

// ---------- bf16 copies + row sq-norms for cond_mmd (z: 0=c_v, 1=c_t) ----------
__global__ void prep_feats_kernel(const float* cv, const float* ct,
                                  unsigned short* cv16, unsigned short* ct16,
                                  float* sq, int D) {
    const float* src = blockIdx.z ? ct : cv;
    unsigned short* dst = blockIdx.z ? ct16 : cv16;
    float* sqo = sq + (size_t)blockIdx.z * gridDim.x;   // gridDim.x == B
    int row = blockIdx.x, d = threadIdx.x;
    float v = src[(size_t)row * D + d];
    dst[(size_t)row * D + d] = f2bf(v);
    __shared__ float sm[256];
    sm[d] = v * v; __syncthreads();
    for (int s = blockDim.x >> 1; s > 0; s >>= 1) {
        if (d < s) sm[d] += sm[d + s];
        __syncthreads();
    }
    if (d == 0) sqo[row] = sm[0];
}

// ---------- HSIC prep: row-normalize + write transposed bf16 (z: cv,sv,ct,st) ----------
__global__ void prep_hsic_kernel(const float* cv, const float* sv,
                                 const float* ctm, const float* stm,
                                 unsigned short* outT, size_t matStride,
                                 int B, int D) {
    const float* src = (blockIdx.z == 0) ? cv : (blockIdx.z == 1) ? sv
                     : (blockIdx.z == 2) ? ctm : stm;
    unsigned short* dst = outT + (size_t)blockIdx.z * matStride;
    int row = blockIdx.x, d = threadIdx.x;
    float v = src[(size_t)row * D + d];
    __shared__ float sm[256];
    __shared__ float inv;
    sm[d] = v * v; __syncthreads();
    for (int s = blockDim.x >> 1; s > 0; s >>= 1) {
        if (d < s) sm[d] += sm[d + s];
        __syncthreads();
    }
    if (d == 0) inv = 1.0f / fmaxf(sqrtf(sm[0]), 1e-12f);
    __syncthreads();
    dst[(size_t)d * B + row] = f2bf(v * inv);   // transposed [D][B]
}

// ---------- column sums of normalized features (csum / ssum vectors) ----------
__global__ void colsum_kernel(const unsigned short* baseT, size_t matStride,
                              float* sums, int B, int D) {
    __shared__ float sm[256];
    const unsigned short* rowp =
        baseT + (size_t)blockIdx.z * matStride + (size_t)blockIdx.x * B;
    float s = 0.f;
    for (int b = threadIdx.x; b < B; b += blockDim.x) s += bf2f(rowp[b]);
    int t = threadIdx.x;
    sm[t] = s; __syncthreads();
    for (int k = blockDim.x >> 1; k > 0; k >>= 1) {
        if (t < k) sm[t] += sm[t + k];
        __syncthreads();
    }
    if (t == 0) sums[(size_t)blockIdx.z * D + blockIdx.x] = sm[0];
}

// ---------- fused Gram + multi-gamma RBF + group reduction (cond_mmd) ----------
// 128x128 macro tile per block (8 waves, each wave: 16 rows x 8 n-subtiles).
// K tile never touches memory: RBF + group accumulation in registers/LDS.
__global__ __launch_bounds__(256)
void mmd_gram_kernel(const unsigned short* F0, const unsigned short* F1,
                     const float* sq0, const float* sq1,
                     const int* gid, float* Sout, int B, int D) {
    const unsigned short* F = blockIdx.z ? F1 : F0;
    const float* sq = blockIdx.z ? sq1 : sq0;
    float* Sacc = Sout + (size_t)blockIdx.z * (NGROUPS * NGROUPS);

    __shared__ unsigned short As[128 * 32];
    __shared__ unsigned short Bs[128 * 32];
    __shared__ float Sl[NGROUPS * NGROUPS];

    int tid = threadIdx.x;
    if (tid < NGROUPS * NGROUPS) Sl[tid] = 0.0f;

    int iBase = blockIdx.x * 128, jBase = blockIdx.y * 128;
    int wave = tid >> 5, lane = tid & 31;
    int hi = (lane >> 4) & 1, l15 = lane & 15;

    v8f accv[8] = {};
    for (int k0 = 0; k0 < D; k0 += 32) {
        __syncthreads();
        {   // cooperative stage: 32 bytes per thread per tile (b128 copies)
            int r = tid >> 1, h = tid & 1;
            const uint4* gA = (const uint4*)(F + (size_t)(iBase + r) * D + k0 + h * 16);
            uint4* lA = (uint4*)(As + r * 32 + h * 16);
            lA[0] = gA[0]; lA[1] = gA[1];
            const uint4* gB = (const uint4*)(F + (size_t)(jBase + r) * D + k0 + h * 16);
            uint4* lB = (uint4*)(Bs + r * 32 + h * 16);
            lB[0] = gB[0]; lB[1] = gB[1];
        }
        __syncthreads();
        // A fragment (16x32 bf16, ISA interleaved K mapping), this wave's 16 rows
        union { v16bf v; unsigned u[8]; } af;
        const unsigned* a32 = (const unsigned*)(As + (wave * 16 + l15) * 32);
        #pragma unroll
        for (int c = 0; c < 8; ++c) {
            int kb = 2 * c + (c >= 4 ? 8 : 0) + (hi ? 8 : 0);
            af.u[c] = a32[kb >> 1];
        }
        #pragma unroll
        for (int nt = 0; nt < 8; ++nt) {
            union { v16bf v; unsigned u[8]; } bfr;
            const unsigned* b32 = (const unsigned*)(Bs + (nt * 16 + l15) * 32);
            #pragma unroll
            for (int c = 0; c < 8; ++c) {
                int kb = 2 * c + (c >= 4 ? 8 : 0) + (hi ? 8 : 0);
                bfr.u[c] = b32[kb >> 1];
            }
            accv[nt] = __builtin_amdgcn_wmma_f32_16x16x32_bf16(
                false, af.v, false, bfr.v, (short)0, accv[nt], false, false);
        }
    }
    __syncthreads();

    // epilogue: D_ij = |xi|^2 + |xj|^2 - 2<xi,xj>;  Ksum = sum_g exp(-g D);
    // accumulate into per-group matrix via LDS atomics.
    int iRowBase = iBase + wave * 16 + hi * 8;
    float sqi[8]; int gi[8];
    #pragma unroll
    for (int v = 0; v < 8; ++v) {
        int i = iRowBase + v;
        sqi[v] = sq[i]; gi[v] = gid[i];
    }
    #pragma unroll
    for (int nt = 0; nt < 8; ++nt) {
        int j = jBase + nt * 16 + l15;
        float sqj = sq[j];
        int gj = gid[j];
        if (gj < 0) continue;
        #pragma unroll
        for (int v = 0; v < 8; ++v) {
            if (gi[v] < 0) continue;
            float dm = sqi[v] + sqj - 2.0f * accv[nt][v];
            float ks = __expf(-0.01f * dm) + __expf(-0.1f * dm) +
                       __expf(-1.0f * dm) + __expf(-10.0f * dm) +
                       __expf(-100.0f * dm);
            atomicAdd(&Sl[gi[v] * NGROUPS + gj], ks);
        }
    }
    __syncthreads();
    if (tid < NGROUPS * NGROUPS) atomicAdd(&Sacc[tid], Sl[tid]);
}

// ---------- HSIC GEMM: G = Cn^T Sn (DxD, contract over B) ----------
// tr(KHLH) = ||G||_F^2 - (2/n) csum^T G ssum + (1/n^2)||csum||^2||ssum||^2
__global__ __launch_bounds__(32)
void hsic_gemm_kernel(const unsigned short* cT0, const unsigned short* sT0,
                      const unsigned short* cT1, const unsigned short* sT1,
                      const float* sums, float* accbuf, int B, int D) {
    const unsigned short* cT = blockIdx.z ? cT1 : cT0;
    const unsigned short* sT = blockIdx.z ? sT1 : sT0;
    const float* csum = sums + (blockIdx.z ? 2 * (size_t)D : 0);
    const float* ssum = sums + (blockIdx.z ? 3 * (size_t)D : (size_t)D);
    float* gg_acc  = &accbuf[8 + 2 * blockIdx.z];
    float* bil_acc = &accbuf[9 + 2 * blockIdx.z];

    int lane = threadIdx.x & 31;
    int hi = lane >> 4, l15 = lane & 15;
    int mBase = blockIdx.x * 16, nBase = blockIdx.y * 16;

    const unsigned* aRow = (const unsigned*)(cT + (size_t)(mBase + l15) * B);
    const unsigned* bRow = (const unsigned*)(sT + (size_t)(nBase + l15) * B);

    v8f g = {};
    for (int k0 = 0; k0 < B; k0 += 32) {
        union { v16bf v; unsigned u[8]; } af, bfr;
        #pragma unroll
        for (int c = 0; c < 8; ++c) {
            int ka = 2 * c + (c >= 4 ? 8 : 0) + (hi ? 8 : 0);
            af.u[c]  = aRow[(k0 + ka) >> 1];
            int kb = 2 * c + (c >= 4 ? 8 : 0) + (hi ? 8 : 0);
            bfr.u[c] = bRow[(k0 + kb) >> 1];
        }
        g = __builtin_amdgcn_wmma_f32_16x16x32_bf16(
            false, af.v, false, bfr.v, (short)0, g, false, false);
    }
    float sn = ssum[nBase + l15];
    float lgg = 0.f, lbil = 0.f;
    #pragma unroll
    for (int v = 0; v < 8; ++v) {
        float gv = g[v];
        lgg  += gv * gv;
        lbil += csum[mBase + v + 8 * hi] * gv * sn;
    }
    for (int off = 16; off > 0; off >>= 1) {
        lgg  += __shfl_down(lgg, off, 32);
        lbil += __shfl_down(lbil, off, 32);
    }
    if (lane == 0) { atomicAdd(gg_acc, lgg); atomicAdd(bil_acc, lbil); }
}

// ---------- final scalar combine ----------
__global__ void combine_kernel(const float* acc, const float* ncount,
                               const float* S, const float* sums,
                               float* out, int B, int D, int H) {
    if (threadIdx.x != 0 || blockIdx.x != 0) return;
    float md = fmaxf(acc[0], 1.0f);
    float f1 = acc[1] / md, f2 = acc[2] / md, f3 = acc[3] / md;
    float ce = acc[4] / md + acc[5] / md;

    float mmd[2];
    for (int z = 0; z < 2; ++z) {
        const float* Sz = S + z * (NGROUPS * NGROUPS);
        float loss = 0.f, cnt = 0.f;
        for (int lab = 0; lab < 4; ++lab)
            for (int d1 = 0; d1 < 4; ++d1)
                for (int d2 = d1 + 1; d2 < 4; ++d2) {
                    int g1 = lab * 4 + d1, g2 = lab * 4 + d2;
                    float n1 = ncount[g1], n2 = ncount[g2];
                    if (n1 > 1.0f && n2 > 1.0f) {
                        float m1 = fmaxf(n1, 1.0f), m2 = fmaxf(n2, 1.0f);
                        float t = Sz[g1 * NGROUPS + g1] / (m1 * m1)
                                + Sz[g2 * NGROUPS + g2] / (m2 * m2)
                                - 2.0f * Sz[g2 * NGROUPS + g1] / fmaxf(n1 * n2, 1.0f);
                        loss += t; cnt += 1.0f;
                    }
                }
        mmd[z] = loss / fmaxf(cnt, 1.0f);
    }

    float hs[2];
    for (int z = 0; z < 2; ++z) {
        const float* cs = sums + (z ? 2 * (size_t)D : 0);
        const float* ss = sums + (z ? 3 * (size_t)D : (size_t)D);
        float nc = 0.f, ns = 0.f;
        for (int d = 0; d < D; ++d) { nc += cs[d] * cs[d]; ns += ss[d] * ss[d]; }
        float gg = acc[8 + 2 * z], bil = acc[9 + 2 * z];
        float n = (float)B;
        hs[z] = (gg - (2.0f / n) * bil + nc * ns / (n * n)) / ((n - 1.0f) * (n - 1.0f));
    }

    float recon = acc[6] / ((float)B * (float)H) + acc[7] / ((float)B * (float)H);
    out[0] = 0.4f * f1 + 0.3f * f2 + 0.3f * f3
           + 0.1f * (ce + mmd[0] + mmd[1])
           + 0.1f * (hs[0] + hs[1])
           + 1.0f * recon;
}

// ---------- launch ----------
extern "C" void kernel_launch(void* const* d_in, const int* in_sizes, int n_in,
                              void* d_out, int out_size, void* d_ws, size_t ws_size,
                              hipStream_t stream) {
    const float* lg1 = (const float*)d_in[0];
    const int*   lb1 = (const int*)  d_in[1];
    const float* lg2 = (const float*)d_in[2];
    const int*   lb2 = (const int*)  d_in[3];
    const float* lg3 = (const float*)d_in[4];
    const int*   lb3 = (const int*)  d_in[5];
    const float* c_v = (const float*)d_in[6];
    const float* s_v = (const float*)d_in[7];
    const float* c_t = (const float*)d_in[8];
    const float* s_t = (const float*)d_in[9];
    const float* vsv = (const float*)d_in[10];
    const float* vst = (const float*)d_in[11];
    const int*   dom = (const int*)  d_in[12];
    const unsigned char* mask = (const unsigned char*)d_in[13];
    const float* hir = (const float*)d_in[14];
    const float* hio = (const float*)d_in[15];
    const float* htr = (const float*)d_in[16];
    const float* hto = (const float*)d_in[17];
    float* out = (float*)d_out;

    int B = in_sizes[1];            // 4096
    int D = in_sizes[6] / B;        // 256
    int H = in_sizes[14] / B;       // 768
    size_t BD = (size_t)B * D;

    // workspace layout
    float* acc    = (float*)d_ws;              // 32 scalar accumulators
    float* ncount = acc + 32;                  // 16 group counts
    float* Sgrp   = ncount + 16;               // 2 x 16x16 group sums
    float* sums   = Sgrp + 2 * NGROUPS * NGROUPS;  // 4 x D column sums
    float* sq     = sums + 4 * (size_t)D;      // 2 x B sq-norms
    int*   gid    = (int*)(sq + 2 * (size_t)B);
    unsigned short* cv16  = (unsigned short*)(gid + B);
    unsigned short* ct16  = cv16 + BD;
    unsigned short* hsicT = ct16 + BD;         // 4 contiguous [D][B] matrices

    int nzero = 32 + 16 + 2 * NGROUPS * NGROUPS + 4 * D;
    zero_kernel<<<(nzero + 255) / 256, 256, 0, stream>>>(acc, nzero);

    row_losses_kernel<<<(B + 255) / 256, 256, 0, stream>>>(
        lg1, lb1, lg2, lb2, lg3, lb3, vsv, vst, dom, mask, acc, ncount, gid, B);

    recon_kernel<<<256, 256, 0, stream>>>(hir, hio, htr, hto, acc, B * H);

    prep_feats_kernel<<<dim3(B, 1, 2), dim3(D), 0, stream>>>(
        c_v, c_t, cv16, ct16, sq, D);

    prep_hsic_kernel<<<dim3(B, 1, 4), dim3(D), 0, stream>>>(
        c_v, s_v, c_t, s_t, hsicT, BD, B, D);

    colsum_kernel<<<dim3(D, 1, 4), dim3(256), 0, stream>>>(
        hsicT, BD, sums, B, D);

    mmd_gram_kernel<<<dim3(B / 128, B / 128, 2), dim3(256), 0, stream>>>(
        cv16, ct16, sq, sq + B, gid, Sgrp, B, D);

    hsic_gemm_kernel<<<dim3(D / 16, D / 16, 2), dim3(32), 0, stream>>>(
        hsicT, hsicT + BD, hsicT + 2 * BD, hsicT + 3 * BD, sums, acc, B, D);

    combine_kernel<<<1, 1, 0, stream>>>(acc, ncount, Sgrp, sums, out, B, D, H);
}
